// Attention_16947940950757
// MI455X (gfx1250) — matrix operand order repaired and checked by reading, not compiled, not run on hardware
//
#include <hip/hip_runtime.h>
#include <hip/hip_bf16.h>

// ---- types ----
typedef __attribute__((ext_vector_type(16))) _Float16 v16h;
typedef __attribute__((ext_vector_type(8)))  _Float16 v8h;
typedef __attribute__((ext_vector_type(8)))  float    v8f;
typedef __attribute__((ext_vector_type(4)))  unsigned int u32x4;
typedef __attribute__((ext_vector_type(8)))  int          i32x8;
typedef __attribute__((ext_vector_type(4)))  int          i32x4;

#define WMMA_F16(a, b, c) \
  __builtin_amdgcn_wmma_f32_16x16x32_f16(false, (a), false, (b), (short)0, (c), false, false)

// Problem constants (B=4, N=2048, DIM=1024, H=16, DH=64)
static constexpr int   kN     = 2048;
static constexpr float kScale = 0.125f;   // 64^-0.5

// Tensor Data Mover availability (device pass only; host pass sees 0)
#if defined(__gfx1250__) && __has_builtin(__builtin_amdgcn_tensor_load_to_lds) && \
    __has_builtin(__builtin_amdgcn_s_wait_tensorcnt)
#define HAVE_TDM 1
#else
#define HAVE_TDM 0
#endif

// ---------------------------------------------------------------------------
// Fragment loaders (CDNA5 wave32 WMMA VGPR layouts, 16-bit data, K=32)
// A 16x32: lanes 0-15 row=lane, K={0..7,16..23}; lanes 16-31 row=lane-16, K={8..15,24..31}
// B 32x16 (from B^T row-major): lanes 0-15 col=lane, K={0..7,16..23}; lanes 16-31 col=lane-16, K={8..15,24..31}
// ---------------------------------------------------------------------------
__device__ __forceinline__ v16h load_a_frag(const _Float16* base, int ldm, int lane) {
  const int r  = lane & 15;
  const int hi = (lane >> 4) & 1;
  const _Float16* p = base + (size_t)r * ldm + hi * 8;
  v8h lo = *(const v8h*)(p);
  v8h up = *(const v8h*)(p + 16);
  v16h a;
#pragma unroll
  for (int i = 0; i < 8; ++i) { a[i] = lo[i]; a[i + 8] = up[i]; }
  return a;
}

__device__ __forceinline__ v16h load_b_frag(const _Float16* base, int ldm, int lane) {
  const int c  = lane & 15;
  const int hi = (lane >> 4) & 1;
  const _Float16* p = base + (size_t)c * ldm + hi * 8;
  v8h lo = *(const v8h*)(p);
  v8h up = *(const v8h*)(p + 16);
  v16h b;
#pragma unroll
  for (int i = 0; i < 8; ++i) { b[i] = lo[i]; b[i + 8] = up[i]; }
  return b;
}

// ---------------------------------------------------------------------------
// TDM: async-load one 16x32 f16 A-slice (row stride 1024 elems) into LDS with
// 16B padding after every 64B -> LDS row stride 80B (bank-conflict-free
// ds_load_b128 fragment reads: start banks 20*r mod 64).
// D# layout per CDNA5 ISA (6-arg builtin form).
// ---------------------------------------------------------------------------
#if HAVE_TDM
__device__ __forceinline__ void tdm_load_a16x32(const _Float16* gsrc, unsigned lds_byte) {
  const unsigned long long ga = (unsigned long long)(size_t)gsrc;
  u32x4 g0;
  g0[0] = 1u;                                               // count=1 (user D#)
  g0[1] = lds_byte;                                         // lds_addr
  g0[2] = (unsigned)ga;                                     // global_addr[31:0]
  g0[3] = (unsigned)((ga >> 32) & 0x01FFFFFFu) | (2u << 30); // global_addr[56:32], type=2
  i32x8 g1;
  g1[0] = (int)((1u << 16)    // data_size = 2 bytes
              | (1u << 20)    // pad_enable
              | (3u << 22)    // pad_interval: 16 dwords (64B)
              | (3u << 25));  // pad_amount: 4 dwords (16B)
  g1[1] = (int)(1024u << 16); // tensor_dim0[15:0] = 1024
  g1[2] = (int)(16u << 16);   // tensor_dim0 hi=0 | tensor_dim1[15:0] = 16
  g1[3] = (int)(32u << 16);   // tensor_dim1 hi=0 | tile_dim0 = 32
  g1[4] = 16;                 // tile_dim1 = 16, tile_dim2 = 0
  g1[5] = 1024;               // tensor_dim0_stride[31:0] = 1024
  g1[6] = 0;
  g1[7] = 0;
  const i32x4 gz4 = {0, 0, 0, 0};
  const i32x8 gz8 = {0, 0, 0, 0, 0, 0, 0, 0};
  __builtin_amdgcn_tensor_load_to_lds(g0, g1, gz4, gz4, gz8, 0);
}
#endif

// Load the 4 B-fragments of a 64-wide column group at depth k.
__device__ __forceinline__ void load_b4(const _Float16* __restrict__ Bt, int k,
                                        int lane, v16h* dst) {
#pragma unroll
  for (int ct = 0; ct < 4; ++ct)
    dst[ct] = load_b_frag(Bt + (size_t)(ct * 16) * 1024 + k, 1024, lane);
}

// ---------------------------------------------------------------------------
// Pipelined 16x64 GEMM tile, K=1024, fully unrolled (16 double-steps) so there
// is no loop backedge: every fragment is a fresh SSA value and the allocator
// can rename freely (no bq rotations => no WMMA->VALU WAR hazard NOPs).
// A via TDM->LDS double buffer (fallback: direct global).
// ---------------------------------------------------------------------------
__device__ __forceinline__ void gemm_tile_16x64(const _Float16* __restrict__ Arow,
                                                const _Float16* __restrict__ Bt,
                                                _Float16* ldsA,   // 1280 halfs per wave
                                                int lane, v8f* acc) {
  (void)ldsA;
#if HAVE_TDM
  const unsigned ldsBase = (unsigned)(size_t)(void*)ldsA;
  tdm_load_a16x32(Arow, ldsBase);                 // slice k=0 -> buffer 0
#endif
  v16h bqA[4], bqB[4];
  load_b4(Bt, 0, lane, bqA);

#pragma unroll
  for (int k0 = 0; k0 < 1024; k0 += 64) {
    // ---- half 1: compute k0 (buf0/bqA); prefetch k0+32 (buf1/bqB)
#if HAVE_TDM
    tdm_load_a16x32(Arow + k0 + 32, ldsBase + 1280u);
#endif
    load_b4(Bt, k0 + 32, lane, bqB);
#if HAVE_TDM
    __builtin_amdgcn_s_wait_tensorcnt(1);          // in-order TDM: slice k0 landed
    __asm__ volatile("" ::: "memory");
    v16h a0 = load_a_frag(ldsA, 40, lane);         // buffer 0, 80B LDS row stride
#else
    v16h a0 = load_a_frag(Arow + k0, 1024, lane);
#endif
#pragma unroll
    for (int ct = 0; ct < 4; ++ct)
      acc[ct] = WMMA_F16(a0, bqA[ct], acc[ct]);

    // ---- half 2: compute k0+32 (buf1/bqB); prefetch k0+64 (buf0/bqA)
    const int kn = (k0 + 64 < 1024) ? (k0 + 64) : 960;   // tail: harmless redundant
#if HAVE_TDM
    tdm_load_a16x32(Arow + kn, ldsBase);
#endif
    load_b4(Bt, kn, lane, bqA);
#if HAVE_TDM
    __builtin_amdgcn_s_wait_tensorcnt(1);          // slice k0+32 landed
    __asm__ volatile("" ::: "memory");
    v16h a1 = load_a_frag(ldsA + 640, 40, lane);   // buffer 1
#else
    v16h a1 = load_a_frag(Arow + k0 + 32, 1024, lane);
#endif
#pragma unroll
    for (int ct = 0; ct < 4; ++ct)
      acc[ct] = WMMA_F16(a1, bqB[ct], acc[ct]);
  }
}

// ---------------------------------------------------------------------------
// Conversion / transpose pre-passes
// ---------------------------------------------------------------------------
__global__ void __launch_bounds__(256)
cvt_f32_to_f16(const float* __restrict__ src, _Float16* __restrict__ dst, int n) {
  int i = blockIdx.x * 256 + threadIdx.x;
  if (i < n) dst[i] = (_Float16)src[i];
}

__global__ void __launch_bounds__(256)
transpose_1024(const float* __restrict__ W, _Float16* __restrict__ Wt) {
  int i  = blockIdx.x * 256 + threadIdx.x;   // i = n*1024 + k
  int nn = i >> 10;
  int k  = i & 1023;
  Wt[i] = (_Float16)W[(size_t)k * 1024 + nn];
}

// ---------------------------------------------------------------------------
// Fused QKV projection: Xh[8192,1024] @ [Wq|Wk|Wv]^T -> Qh/Kh [B,H,N,64], Vt [B,H,64,N]
// ---------------------------------------------------------------------------
__global__ void __launch_bounds__(256)
qkv_gemm(const _Float16* __restrict__ Xh, const _Float16* __restrict__ WtA,
         _Float16* __restrict__ Qh, _Float16* __restrict__ Kh, _Float16* __restrict__ Vt) {
  __shared__ _Float16 aStage[8 * 1280];      // 2 TDM buffers x 640 halfs per wave
  const int lane = threadIdx.x & 31;
  const int wave = threadIdx.x >> 5;
  const int task = blockIdx.x * 8 + wave;    // 0..24575
  const int mt   = task & 511;
  const int ng   = task >> 9;                // 0..47
  const int m0   = mt * 16;
  const int n0   = ng * 64;

  v8f acc[4] = {};
  gemm_tile_16x64(Xh + (size_t)m0 * 1024, WtA + (size_t)n0 * 1024,
                  aStage + wave * 1280, lane, acc);

  const int cl     = lane & 15;
  const int rowadd = ((lane >> 4) & 1) * 8;
#pragma unroll
  for (int ct = 0; ct < 4; ++ct) {
    const int ccol  = n0 + ct * 16 + cl;   // 0..3071
    const int which = ccol >> 10;          // 0=Q 1=K 2=V
    const int inner = ccol & 1023;
    const int h = inner >> 6, d = inner & 63;
#pragma unroll
    for (int r = 0; r < 8; ++r) {
      const int m  = m0 + r + rowadd;
      const int b_ = m >> 11, n = m & 2047;
      const size_t bh = (size_t)(b_ * 16 + h);
      const _Float16 val = (_Float16)acc[ct][r];
      if (which == 0)      Qh[(bh * kN + n) * 64 + d] = val;
      else if (which == 1) Kh[(bh * kN + n) * 64 + d] = val;
      else                 Vt[(bh * 64 + d) * kN + n] = val;
    }
  }
}

// ---------------------------------------------------------------------------
// One 32-key flash-attention step: S-wmmas from kf, prefetch next K block
// into kn_out (different registers => no rotation copies), V fragments issued
// before the softmax VALU work, P relayout via per-wave LDS buffer.
// ---------------------------------------------------------------------------
__device__ __forceinline__ void attn_step(int j0, int jn,
                                          const v16h* kf, v16h* kn_out,
                                          const v16h& aq0, const v16h& aq1,
                                          const _Float16* __restrict__ kbase,
                                          const _Float16* __restrict__ vbase,
                                          _Float16* pbuf, int lane, int cl, int rowadd,
                                          float* mrow, float* lrow, v8f* o) {
  v8f s0 = {}, s1 = {};
  s0 = WMMA_F16(aq0, kf[0], s0);
  s0 = WMMA_F16(aq1, kf[1], s0);
  s1 = WMMA_F16(aq0, kf[2], s1);
  s1 = WMMA_F16(aq1, kf[3], s1);

  // prefetch next K block + current V block (overlap softmax VALU)
#pragma unroll
  for (int t = 0; t < 2; ++t) {
    kn_out[2 * t + 0] = load_b_frag(kbase + (size_t)(jn + 16 * t) * 64, 64, lane);
    kn_out[2 * t + 1] = load_b_frag(kbase + (size_t)(jn + 16 * t) * 64 + 32, 64, lane);
  }
  v16h vb[4];
#pragma unroll
  for (int n = 0; n < 4; ++n)
    vb[n] = load_b_frag(vbase + (size_t)(n * 16) * kN + j0, kN, lane);

  // online softmax per row (row = r + rowadd); cols across 16 lanes
#pragma unroll
  for (int r = 0; r < 8; ++r) {
    float t0 = s0[r] * kScale;
    float t1 = s1[r] * kScale;
    float mx = fmaxf(t0, t1);
    mx = fmaxf(mx, __shfl_xor(mx, 1, 32));
    mx = fmaxf(mx, __shfl_xor(mx, 2, 32));
    mx = fmaxf(mx, __shfl_xor(mx, 4, 32));
    mx = fmaxf(mx, __shfl_xor(mx, 8, 32));
    const float mnew  = fmaxf(mrow[r], mx);
    const float alpha = __expf(mrow[r] - mnew);
    const float p0 = __expf(t0 - mnew);
    const float p1 = __expf(t1 - mnew);
    float sum = p0 + p1;
    sum += __shfl_xor(sum, 1, 32);
    sum += __shfl_xor(sum, 2, 32);
    sum += __shfl_xor(sum, 4, 32);
    sum += __shfl_xor(sum, 8, 32);
    lrow[r] = lrow[r] * alpha + sum;
    mrow[r] = mnew;

    const int row = r + rowadd;
    pbuf[row * 32 + cl]      = (_Float16)p0;
    pbuf[row * 32 + 16 + cl] = (_Float16)p1;
#pragma unroll
    for (int n = 0; n < 4; ++n) o[n][r] *= alpha;
  }

  // LDS in-order per wave; fence compiler + wait DS writes before re-read
  __asm__ volatile("s_wait_dscnt 0" ::: "memory");

  v16h pa = load_a_frag(pbuf, 32, lane);
#pragma unroll
  for (int n = 0; n < 4; ++n)
    o[n] = WMMA_F16(pa, vb[n], o[n]);
  __asm__ volatile("" ::: "memory");   // keep pa reads ahead of next-step stores
}

// ---------------------------------------------------------------------------
// Flash attention: one wave = 16 query rows of one (b,h). Key loop manually
// 2x-unrolled with K-fragment register ping-pong (kfA/kfB).
// ---------------------------------------------------------------------------
__global__ void __launch_bounds__(256)
attn_kernel(const _Float16* __restrict__ Qh, const _Float16* __restrict__ Kh,
            const _Float16* __restrict__ Vt, _Float16* __restrict__ attOut) {
  __shared__ _Float16 pshm[8 * 16 * 32];   // 1KB per wave: P relayout buffer
  const int lane = threadIdx.x & 31;
  const int wave = threadIdx.x >> 5;
  _Float16* pbuf = pshm + wave * (16 * 32);

  const int task = blockIdx.x * 8 + wave;  // 0..8191
  const int qt   = task & 127;
  const int bh   = task >> 7;              // 0..63
  const int i0   = qt * 16;

  const _Float16* qbase = Qh + ((size_t)bh * kN + i0) * 64;
  const _Float16* kbase = Kh + (size_t)bh * kN * 64;
  const _Float16* vbase = Vt + (size_t)bh * 64 * kN;

  const v16h aq0 = load_a_frag(qbase, 64, lane);
  const v16h aq1 = load_a_frag(qbase + 32, 64, lane);

  float mrow[8], lrow[8];
  v8f o[4] = {};
#pragma unroll
  for (int r = 0; r < 8; ++r) { mrow[r] = -3.0e38f; lrow[r] = 0.0f; }

  const int cl     = lane & 15;
  const int rowadd = ((lane >> 4) & 1) * 8;

  // prologue: K fragments for the first key block
  v16h kfA[4], kfB[4];
#pragma unroll
  for (int t = 0; t < 2; ++t) {
    kfA[2 * t + 0] = load_b_frag(kbase + (size_t)(16 * t) * 64, 64, lane);
    kfA[2 * t + 1] = load_b_frag(kbase + (size_t)(16 * t) * 64 + 32, 64, lane);
  }

  for (int j0 = 0; j0 < kN; j0 += 64) {
    attn_step(j0, j0 + 32, kfA, kfB, aq0, aq1, kbase, vbase,
              pbuf, lane, cl, rowadd, mrow, lrow, o);
    const int jn2 = (j0 + 64 < kN) ? (j0 + 64) : (j0 + 32);  // tail redundant
    attn_step(j0 + 32, jn2, kfB, kfA, aq0, aq1, kbase, vbase,
              pbuf, lane, cl, rowadd, mrow, lrow, o);
  }

  // --- normalize and store [B, N, H*64] f16 for the output projection
  const int b_ = bh >> 4, h = bh & 15;
#pragma unroll
  for (int r = 0; r < 8; ++r) {
    const float inv = 1.0f / lrow[r];
    const int row   = i0 + r + rowadd;
    const size_t base = ((size_t)(b_ * kN + row)) * 1024 + h * 64 + cl;
#pragma unroll
    for (int n = 0; n < 4; ++n)
      attOut[base + (size_t)n * 16] = (_Float16)(o[n][r] * inv);
  }
}

// ---------------------------------------------------------------------------
// Output projection: AttOut[8192,1024] @ Wo + bo -> f32 out
// ---------------------------------------------------------------------------
__global__ void __launch_bounds__(256)
out_proj(const _Float16* __restrict__ Ao, const _Float16* __restrict__ WoT,
         const float* __restrict__ bo, float* __restrict__ out) {
  __shared__ _Float16 aStage[8 * 1280];
  const int lane = threadIdx.x & 31;
  const int wave = threadIdx.x >> 5;
  const int task = blockIdx.x * 8 + wave;  // 0..8191
  const int mt   = task & 511;
  const int cg   = task >> 9;              // 0..15
  const int m0   = mt * 16;
  const int c0   = cg * 64;

  v8f acc[4] = {};
  gemm_tile_16x64(Ao + (size_t)m0 * 1024, WoT + (size_t)c0 * 1024,
                  aStage + wave * 1280, lane, acc);

  const int cl     = lane & 15;
  const int rowadd = ((lane >> 4) & 1) * 8;
#pragma unroll
  for (int ct = 0; ct < 4; ++ct) {
    const int col = c0 + ct * 16 + cl;
    const float bias = bo[col];
#pragma unroll
    for (int r = 0; r < 8; ++r) {
      const int m = m0 + r + rowadd;
      out[(size_t)m * 1024 + col] = acc[ct][r] + bias;
    }
  }
}

// ---------------------------------------------------------------------------
// Host launcher
// ---------------------------------------------------------------------------
extern "C" void kernel_launch(void* const* d_in, const int* in_sizes, int n_in,
                              void* d_out, int out_size, void* d_ws, size_t ws_size,
                              hipStream_t stream) {
  const float* x  = (const float*)d_in[0];
  const float* Wq = (const float*)d_in[1];
  const float* Wk = (const float*)d_in[2];
  const float* Wv = (const float*)d_in[3];
  const float* Wo = (const float*)d_in[4];
  const float* bo = (const float*)d_in[5];
  float* out = (float*)d_out;

  char* ws = (char*)d_ws;                               // 88 MB used
  _Float16* Xh  = (_Float16*)(ws);                      // 16 MB  x in f16
  _Float16* WtA = (_Float16*)(ws + (16ull << 20));      //  6 MB  [Wq|Wk|Wv]^T f16
  _Float16* WoT = (_Float16*)(ws + (22ull << 20));      //  2 MB  Wo^T f16
  _Float16* Qh  = (_Float16*)(ws + (24ull << 20));      // 16 MB  [B,H,N,64]
  _Float16* Kh  = (_Float16*)(ws + (40ull << 20));      // 16 MB  [B,H,N,64]
  _Float16* Vt  = (_Float16*)(ws + (56ull << 20));      // 16 MB  [B,H,64,N]
  _Float16* Ao  = (_Float16*)(ws + (72ull << 20));      // 16 MB  [B,N,1024]

  const int nx = 4 * 2048 * 1024;
  cvt_f32_to_f16<<<nx / 256, 256, 0, stream>>>(x, Xh, nx);
  transpose_1024<<<4096, 256, 0, stream>>>(Wq, WtA);
  transpose_1024<<<4096, 256, 0, stream>>>(Wk, WtA + (1u << 20));
  transpose_1024<<<4096, 256, 0, stream>>>(Wv, WtA + (2u << 20));
  transpose_1024<<<4096, 256, 0, stream>>>(Wo, WoT);

  qkv_gemm<<<3072, 256, 0, stream>>>(Xh, WtA, Qh, Kh, Vt);
  attn_kernel<<<1024, 256, 0, stream>>>(Qh, Kh, Vt, Ao);
  out_proj<<<1024, 256, 0, stream>>>(Ao, WoT, bo, out);
}
